// RecurrentLFQ_31482110279875
// MI455X (gfx1250) — compile-verified, matrix operand order checked by reading, not run on hardware
//
#include <hip/hip_runtime.h>
#include <hip/hip_bf16.h>

typedef __attribute__((ext_vector_type(16))) _Float16 v16h;
typedef __attribute__((ext_vector_type(8)))  float    v8f;

#define TOKENS      16384
#define D           512
#define NSTEP       12
#define OT_PER_GATE 32          // 512 cols / 16 per WMMA tile
#define KC          16          // 512 K / 32 per WMMA
#define OUT_ELEMS   (TOKENS*D)  // 8,388,608 floats of `out`
#define IDX_OFF     OUT_ELEMS
#define LOG_EPS     -11.512925465f

// K offset within a 32-wide K chunk for f16 A/B fragment half j, per the
// CDNA5 ISA 16-bit 16x32 A-matrix table (lanes>=16 hold K+8; VGPR4..7 hold K+16).
__device__ __forceinline__ int kmap(int j, int hi8) {
  int v = j >> 1, p = j & 1;
  return hi8 + ((v < 4) ? 0 : 16) + ((v & 3) << 1) + p;
}

// ---------------------------------------------------------------------------
// Pack W_hh (1536x512 f32, row-major, gh = h0 @ W_hh^T) into f16 WMMA
// B-fragments: layout [o_tile(96)][k_chunk(16)][lane(32)][16 halves].
// B element (K, N): N = o column = o_tile*16 + (lane&15); K per kmap().
// ---------------------------------------------------------------------------
__global__ __launch_bounds__(256) void pack_whh(const float* __restrict__ W_hh,
                                                _Float16* __restrict__ wfrag) {
  int id = blockIdx.x * blockDim.x + threadIdx.x;  // fragment-lane slot
  if (id >= 96 * KC * 32) return;
  int lane = id & 31;
  int kc   = (id >> 5) & 15;
  int ot   = id >> 9;                              // 0..95 over 1536 columns
  int hi8  = (lane >= 16) ? 8 : 0;
  int ofull = ot * 16 + (lane & 15);
  v16h frag;
#pragma unroll
  for (int j = 0; j < 16; ++j) {
    int K = kc * 32 + kmap(j, hi8);
    frag[j] = (_Float16)W_hh[ofull * D + K];
  }
  *(v16h*)(wfrag + (size_t)id * 16) = frag;
}

// ---------------------------------------------------------------------------
// Main recurrent kernel: 4 waves / WG, 16 tokens per wave, h resident in LDS.
// ---------------------------------------------------------------------------
__global__ __launch_bounds__(128) void rbq_main(
    const float* __restrict__ x,    const float* __restrict__ W_ih,
    const float* __restrict__ b_ih, const float* __restrict__ b_hh,
    const float* __restrict__ W_out,const float* __restrict__ b_out,
    const float* __restrict__ W_po, const float* __restrict__ b_po,
    const _Float16* __restrict__ wfrag,
    float* __restrict__ outs_ws, float* __restrict__ dout)
{
  extern __shared__ float sm[];
  float* h_lds   = sm;                 // 64*512 f32 = 128 KB
  float* inp_lds = sm + 64 * D;        // 64
  float* red     = inp_lds + 64;       // 128
  float* shidxf  = red + 128;          // 64

  int tid  = threadIdx.x;
  int lane = tid & 31;
  int wave = tid >> 5;
  int hi   = lane >> 4;
  int hi8  = hi << 3;
  int m    = lane & 15;
  int ltokA = wave * 16 + m;           // local token row this lane owns
  int wgTokBase = blockIdx.x * 64;
  int gtokA = wgTokBase + ltokA;

  for (int i = tid; i < 64 * D; i += 128) h_lds[i] = 0.f;
  if (tid < 64) inp_lds[tid] = 0.f;
  __syncthreads();

  int idxbits = 0;

#pragma unroll 1
  for (int step = 0; step < NSTEP; ++step) {
    // ---- A fragments: h0 = h + emb, cvt to f16 in ISA A layout ----
    v16h afrag[KC];
#pragma unroll
    for (int kc = 0; kc < KC; ++kc) {
#pragma unroll
      for (int jp = 0; jp < 8; ++jp) {
        int K = kc * 32 + kmap(2 * jp, hi8);     // even K -> 8B aligned pairs
        float2 hv = *(const float2*)&h_lds[ltokA * D + K];
        float2 xv = *(const float2*)&x[(size_t)gtokA * D + K];
        afrag[kc][2 * jp]     = (_Float16)(hv.x + xv.x);
        afrag[kc][2 * jp + 1] = (_Float16)(hv.y + xv.y);
      }
    }
    __syncthreads();

    // ---- 32 column tiles: 3 gates x 16 K chunks of WMMA, then gate math ----
#pragma unroll 1
    for (int ot = 0; ot < OT_PER_GATE; ++ot) {
      int col = (ot << 4) + m;
      float wir = W_ih[col], wiz = W_ih[D + col], win = W_ih[2 * D + col];
      float br_   = b_ih[col] + b_hh[col];            // gx+gh biases fold for r,z
      float bz_   = b_ih[D + col] + b_hh[D + col];
      float bn_ih = b_ih[2 * D + col];                // gx_n bias stays outside r*
      float bn_hh = b_hh[2 * D + col];
      v8f accr, accz, accn;
#pragma unroll
      for (int i = 0; i < 8; ++i) { accr[i] = br_; accz[i] = bz_; accn[i] = bn_hh; }

      const _Float16* wr = wfrag + ((size_t)(( 0 + ot) * KC) * 32 + lane) * 16;
      const _Float16* wz = wfrag + ((size_t)((32 + ot) * KC) * 32 + lane) * 16;
      const _Float16* wn = wfrag + ((size_t)((64 + ot) * KC) * 32 + lane) * 16;
#pragma unroll
      for (int kc = 0; kc < KC; ++kc) {
        v16h brf = *(const v16h*)(wr + (size_t)kc * 32 * 16);
        v16h bzf = *(const v16h*)(wz + (size_t)kc * 32 * 16);
        v16h bnf = *(const v16h*)(wn + (size_t)kc * 32 * 16);
        accr = __builtin_amdgcn_wmma_f32_16x16x32_f16(false, afrag[kc], false, brf,
                                                      (short)0, accr, false, false);
        accz = __builtin_amdgcn_wmma_f32_16x16x32_f16(false, afrag[kc], false, bzf,
                                                      (short)0, accz, false, false);
        accn = __builtin_amdgcn_wmma_f32_16x16x32_f16(false, afrag[kc], false, bnf,
                                                      (short)0, accn, false, false);
      }
      // C/D layout: VGPR v holds row m2 = v + 8*hi, column = lane&15 within tile
#pragma unroll
      for (int v = 0; v < 8; ++v) {
        int m2   = v + hi8;
        int lrow = wave * 16 + m2;
        float iv  = inp_lds[lrow];
        float h0v = h_lds[lrow * D + col] + x[(size_t)(wgTokBase + lrow) * D + col];
        float rr = 1.f / (1.f + __expf(-(accr[v] + iv * wir)));
        float zz = 1.f / (1.f + __expf(-(accz[v] + iv * wiz)));
        float gn = iv * win + bn_ih + rr * accn[v];
        float e2 = __expf(2.f * gn);
        float nn = 1.f - 2.f / (e2 + 1.f);             // overflow-safe tanh
        h_lds[lrow * D + col] = (1.f - zz) * nn + zz * h0v;
      }
    }
    __syncthreads();

    // ---- out = h_new . W_out + b_out (two lanes per token, half K each) ----
    {
      float partial = 0.f;
      int k0 = hi ? (D / 2) : 0;
      const float4* hp = (const float4*)&h_lds[ltokA * D + k0];
      const float4* wp = (const float4*)&W_out[k0];
#pragma unroll 8
      for (int i = 0; i < D / 8; ++i) {
        float4 a = hp[i]; float4 b = wp[i];
        partial += a.x * b.x + a.y * b.y + a.z * b.z + a.w * b.w;
      }
      red[tid] = partial;
    }
    __syncthreads();
    if (lane < 16) {
      float out = red[tid] + red[tid + 16] + b_out[0];
      outs_ws[(size_t)gtokA * NSTEP + step] = out;
      inp_lds[ltokA] = (out > 0.f) ? 1.f : -1.f;       // CODEBOOK_SCALE = 1
      idxbits |= (out > 0.f) ? (1 << (11 - step)) : 0; // MSB-first bit pack
    }
    __syncthreads();
  }

  // ---- indices + project_out (forward x_st == qs) ----
  int* shidx = (int*)shidxf;
  if (lane < 16) {
    shidx[ltokA] = idxbits;
    dout[IDX_OFF + gtokA] = (float)idxbits;
  }
  __syncthreads();
#pragma unroll 1
  for (int mm = 0; mm < 16; ++mm) {
    int bits = shidx[wave * 16 + mm];
    int gtok = wgTokBase + wave * 16 + mm;
    for (int d0 = lane; d0 < D; d0 += 32) {
      float acc = b_po[d0];
#pragma unroll
      for (int j = 0; j < NSTEP; ++j) {
        float q = ((bits >> (11 - j)) & 1) ? 1.f : -1.f;
        acc += q * W_po[j * D + d0];
      }
      dout[(size_t)gtok * D + d0] = acc;
    }
  }
}

// ---------------------------------------------------------------------------
// Entropy pass 1: softmax over 4096 codes factorizes into 12 Bernoullis:
// p_j = prod_d sigma(400*out_d*c_jd). Build 64-entry hi/lo product tables
// per sample in LDS; deterministic per-thread code ownership (16 codes each).
// ---------------------------------------------------------------------------
__global__ __launch_bounds__(256) void entropy_pass1(
    const float* __restrict__ outs_ws, float* __restrict__ prob_part,
    float* __restrict__ ent_part)
{
  __shared__ float fp[12];
  __shared__ float Phi[64], LPhi[64], Plo[64], LPlo[64];
  __shared__ float probsum[4096];
  __shared__ float redu[256];
  int tid = threadIdx.x, wg = blockIdx.x;
  for (int i = tid; i < 4096; i += 256) probsum[i] = 0.f;
  float entAcc = 0.f;
  __syncthreads();

#pragma unroll 1
  for (int s = 0; s < 256; ++s) {
    int tok = wg * 256 + s;
    if (tid < 12) fp[tid] = 1.f / (1.f + __expf(-400.f * outs_ws[tok * 12 + tid]));
    __syncthreads();
    if (tid < 64) {                   // dims 0..5 (MSBs of code index)
      float p = 1.f;
#pragma unroll
      for (int d = 0; d < 6; ++d) {
        int bit = (tid >> (5 - d)) & 1;
        p *= bit ? fp[d] : (1.f - fp[d]);
      }
      Phi[tid] = p; LPhi[tid] = __logf(p);
    } else if (tid < 128) {           // dims 6..11 (LSBs)
      int t = tid - 64;
      float p = 1.f;
#pragma unroll
      for (int d = 6; d < 12; ++d) {
        int bit = (t >> (11 - d)) & 1;
        p *= bit ? fp[d] : (1.f - fp[d]);
      }
      Plo[t] = p; LPlo[t] = __logf(p);
    }
    __syncthreads();
#pragma unroll
    for (int i = 0; i < 16; ++i) {
      int j  = tid * 16 + i;
      int jh = j >> 6, jl = j & 63;
      float p  = Phi[jh] * Plo[jl];
      float lp = (p < 1e-5f) ? LOG_EPS : (LPhi[jh] + LPlo[jl]);
      entAcc -= p * lp;               // -p * log(max(p, eps))
      probsum[j] += p;                // exclusive per-thread ownership of j
    }
    __syncthreads();
  }
  for (int i = tid; i < 4096; i += 256) prob_part[(size_t)wg * 4096 + i] = probsum[i];
  redu[tid] = entAcc;
  __syncthreads();
  for (int o = 128; o > 0; o >>= 1) { if (tid < o) redu[tid] += redu[tid + o]; __syncthreads(); }
  if (tid == 0) ent_part[wg] = redu[0];
}

// ---------------------------------------------------------------------------
// Entropy pass 2: fixed-order reductions -> aux = 0.1*(pse - codebook_entropy)
// ---------------------------------------------------------------------------
__global__ __launch_bounds__(256) void entropy_pass2(
    const float* __restrict__ prob_part, const float* __restrict__ ent_part,
    float* __restrict__ dout, int out_size)
{
  __shared__ float redu[256];
  int tid = threadIdx.x;
  float cAcc = 0.f;
  for (int i = tid; i < 4096; i += 256) {
    float s = 0.f;
    for (int w = 0; w < 64; ++w) s += prob_part[(size_t)w * 4096 + i];
    float avg = s * (1.f / 16384.f);
    cAcc -= avg * __logf(fmaxf(avg, 1e-5f));
  }
  redu[tid] = cAcc;
  __syncthreads();
  for (int o = 128; o > 0; o >>= 1) { if (tid < o) redu[tid] += redu[tid + o]; __syncthreads(); }
  if (tid == 0) {
    float pse = 0.f;
    for (int w = 0; w < 64; ++w) pse += ent_part[w];
    pse *= (1.f / 16384.f);
    dout[out_size - 1] = (pse - redu[0]) * 0.1f;
  }
}

extern "C" void kernel_launch(void* const* d_in, const int* in_sizes, int n_in,
                              void* d_out, int out_size, void* d_ws, size_t ws_size,
                              hipStream_t stream) {
  const float* x     = (const float*)d_in[0];
  const float* W_ih  = (const float*)d_in[1];
  const float* W_hh  = (const float*)d_in[2];
  const float* b_ih  = (const float*)d_in[3];
  const float* b_hh  = (const float*)d_in[4];
  const float* W_out = (const float*)d_in[5];
  const float* b_out = (const float*)d_in[6];
  const float* W_po  = (const float*)d_in[7];
  const float* b_po  = (const float*)d_in[8];
  float* dout = (float*)d_out;

  char* ws = (char*)d_ws;
  _Float16* wfrag  = (_Float16*)ws;                            // 1,572,864 B
  float* outs_ws   = (float*)(ws + 1572864);                   //   786,432 B
  float* prob_part = (float*)(ws + 1572864 + 786432);          // 1,048,576 B
  float* ent_part  = (float*)(ws + 1572864 + 786432 + 1048576);//       256 B

  pack_whh<<<192, 256, 0, stream>>>(W_hh, wfrag);
  size_t smem = (size_t)(64 * D + 64 + 128 + 64) * sizeof(float);  // 132,096 B (320KB WGP LDS)
  rbq_main<<<TOKENS / 64, 128, smem, stream>>>(x, W_ih, b_ih, b_hh, W_out, b_out,
                                               W_po, b_po, wfrag, outs_ws, dout);
  entropy_pass1<<<64, 256, 0, stream>>>(outs_ws, prob_part, ent_part);
  entropy_pass2<<<1, 256, 0, stream>>>(prob_part, ent_part, dout, out_size);
}
